// AttLayer_10857677325046
// MI455X (gfx1250) — matrix operand order repaired and checked by reading, not compiled, not run on hardware
//
#include <hip/hip_runtime.h>
#include <hip/hip_bf16.h>

// ---- types ----------------------------------------------------------------
typedef __attribute__((ext_vector_type(16))) __bf16 v16bf;
typedef __attribute__((ext_vector_type(8)))  __bf16 v8bf;
typedef __attribute__((ext_vector_type(8)))  float  v8f;
typedef __attribute__((ext_vector_type(4)))  float  v4f;

#define B_    32
#define T_    2048
#define D_    512
#define M_TOT (B_ * T_)     // 65536 rows in the flattened GEMM
#define MT_   64            // M-tile per block (4 row-tiles of 16 per wave)
#define EPS_  1e-7f

// ---- K0: W [k][n] fp32 -> Wt [n][k] bf16 (B-fragment friendly) ------------
__global__ __launch_bounds__(256) void wt_convert_kernel(
    const float* __restrict__ W, __bf16* __restrict__ Wt)
{
    int idx = blockIdx.x * 256 + threadIdx.x;     // 512*512 = 262144 launched
    int k = idx >> 9;
    int n = idx & 511;
    Wt[(size_t)n * D_ + k] = (__bf16)W[(size_t)k * D_ + n];
}

// ---- K1: per 64-row tile: C = tanh(x@W + b), ait = C . u ------------------
// 256 threads = 8 waves. Wave w covers N-chunks [w*4 .. w*4+3] (16 cols each)
// for ALL 4 row-tiles, so each B fragment feeds 4 WMMAs.
__global__ __launch_bounds__(256) void gemm_tanh_dot_kernel(
    const float*  __restrict__ x,
    const __bf16* __restrict__ Wt,      // [n][k] bf16
    const float*  __restrict__ bias,
    const float*  __restrict__ u,
    float*        __restrict__ ait)     // [M_TOT]
{
    __shared__ __bf16 At[MT_][528];     // 64 rows x 512 K, padded stride
    __shared__ float  s_ait[MT_];

    const int tid  = threadIdx.x;
    const int lane = tid & 31;
    const int wave = tid >> 5;
    const size_t m0 = (size_t)blockIdx.x * MT_;

    if (tid < MT_) s_ait[tid] = 0.0f;

    // stage x tile (64x512 fp32) into LDS as bf16
    const float* xp = x + m0 * D_;
    #pragma unroll
    for (int i = 0; i < 32; ++i) {
        int f = i * 1024 + tid * 4;               // 32768 floats total
        int r = f >> 9;
        int c = f & 511;
        v4f v = *(const v4f*)(xp + (size_t)r * D_ + c);
        At[r][c + 0] = (__bf16)v[0];
        At[r][c + 1] = (__bf16)v[1];
        At[r][c + 2] = (__bf16)v[2];
        At[r][c + 3] = (__bf16)v[3];
    }
    __syncthreads();

    // ISA 7.12.2 lane mapping
    const int m      = lane & 15;                 // A row (within row-tile)
    const int half   = lane >> 4;                 // 0: lanes 0-15, 1: 16-31
    const int akb    = half ? 8 : 0;              // A lane K-base (per 16-K blk)
    const int bkb    = half ? 8 : 0;              // B lane K-base (per 16-K blk)
    const int n_lane = lane & 15;                 // B/C column

    v8f acc[4][4] = {};                           // [row-tile][N-chunk]

    for (int k0 = 0; k0 < D_; k0 += 32) {
        // 4 A fragments, one per row-tile
        v16bf af[4];
        #pragma unroll
        for (int mt = 0; mt < 4; ++mt) {
            v8bf alo = *(const v8bf*)&At[mt * 16 + m][k0 + akb];
            v8bf ahi = *(const v8bf*)&At[mt * 16 + m][k0 + 16 + akb];
            #pragma unroll
            for (int i = 0; i < 8; ++i) { af[mt][i] = alo[i]; af[mt][i + 8] = ahi[i]; }
        }

        #pragma unroll
        for (int c = 0; c < 4; ++c) {
            int nn = (wave * 4 + c) * 16 + n_lane;
            const __bf16* bp = Wt + (size_t)nn * D_ + k0 + bkb; // 8 K + 8 K at +16
            v8bf blo = *(const v8bf*)(bp);
            v8bf bhi = *(const v8bf*)(bp + 16);
            v16bf bfv;
            #pragma unroll
            for (int i = 0; i < 8; ++i) { bfv[i] = blo[i]; bfv[i + 8] = bhi[i]; }

            #pragma unroll
            for (int mt = 0; mt < 4; ++mt)
                acc[mt][c] = __builtin_amdgcn_wmma_f32_16x16x32_bf16(
                    false, af[mt], false, bfv, (short)0, acc[mt][c], false, false);
        }
    }

    // epilogue: +bias, tanh, * u, accumulate per-row partials
    float racc[4][8];
    #pragma unroll
    for (int mt = 0; mt < 4; ++mt)
        #pragma unroll
        for (int r = 0; r < 8; ++r) racc[mt][r] = 0.0f;

    #pragma unroll
    for (int c = 0; c < 4; ++c) {
        int nn   = (wave * 4 + c) * 16 + n_lane;
        float bv = bias[nn];
        float uv = u[nn];
        #pragma unroll
        for (int mt = 0; mt < 4; ++mt)
            #pragma unroll
            for (int r = 0; r < 8; ++r)            // row = mt*16 + r + 8*half
                racc[mt][r] += tanhf(acc[mt][c][r] + bv) * uv;
    }

    // butterfly reduce over the 16 lanes of each half (cols of C)
    #pragma unroll
    for (int off = 1; off < 16; off <<= 1) {
        #pragma unroll
        for (int mt = 0; mt < 4; ++mt)
            #pragma unroll
            for (int r = 0; r < 8; ++r)
                racc[mt][r] += __shfl_xor(racc[mt][r], off, 32);
    }
    if (n_lane == 0) {
        #pragma unroll
        for (int mt = 0; mt < 4; ++mt)
            #pragma unroll
            for (int r = 0; r < 8; ++r)
                atomicAdd(&s_ait[mt * 16 + r + 8 * half], racc[mt][r]); // ds_add_f32
    }
    __syncthreads();
    if (tid < MT_) ait[m0 + tid] = s_ait[tid];
}

// ---- K2: a = exp(ait) / (sum_T exp(ait) + eps), per batch -----------------
__global__ __launch_bounds__(256) void softmax_t_kernel(
    const float* __restrict__ ait, float* __restrict__ a)
{
    __shared__ float red[8];
    const int b   = blockIdx.x;
    const int tid = threadIdx.x;
    const float* ap = ait + (size_t)b * T_;
    float*       op = a   + (size_t)b * T_;

    float s = 0.0f;
    #pragma unroll
    for (int j = 0; j < 8; ++j) {
        float e = __expf(ap[j * 256 + tid]);
        op[j * 256 + tid] = e;
        s += e;
    }
    #pragma unroll
    for (int off = 1; off < 32; off <<= 1) s += __shfl_xor(s, off, 32);
    if ((tid & 31) == 0) red[tid >> 5] = s;
    __syncthreads();
    if (tid == 0) {
        float t = 0.0f;
        #pragma unroll
        for (int i = 0; i < 8; ++i) t += red[i];
        red[0] = 1.0f / (t + EPS_);
    }
    __syncthreads();
    float inv = red[0];
    #pragma unroll
    for (int j = 0; j < 8; ++j) op[j * 256 + tid] *= inv;
}

// ---- K3: out[b,d] = sum_t x[b,t,d] * a[b,t] -------------------------------
__global__ __launch_bounds__(256) void pool_kernel(
    const float* __restrict__ x, const float* __restrict__ a,
    float* __restrict__ out)
{
    __shared__ float s_a[256];
    const int b = blockIdx.x >> 1;
    const int d = (blockIdx.x & 1) * 256 + threadIdx.x;
    const float* xp = x + (size_t)b * T_ * D_ + d;
    const float* ap = a + (size_t)b * T_;

    float acc = 0.0f;
    for (int tb = 0; tb < T_; tb += 256) {
        __syncthreads();
        s_a[threadIdx.x] = ap[tb + threadIdx.x];
        __syncthreads();
        #pragma unroll 8
        for (int tt = 0; tt < 256; ++tt)
            acc += xp[(size_t)(tb + tt) * D_] * s_a[tt];
    }
    out[(size_t)b * D_ + d] = acc;
}

// ---- launcher -------------------------------------------------------------
extern "C" void kernel_launch(void* const* d_in, const int* in_sizes, int n_in,
                              void* d_out, int out_size, void* d_ws, size_t ws_size,
                              hipStream_t stream)
{
    const float* x    = (const float*)d_in[0];   // [32,2048,512]
    const float* W    = (const float*)d_in[1];   // [512,512]
    const float* bias = (const float*)d_in[2];   // [512]
    const float* u    = (const float*)d_in[3];   // [512]
    float* out = (float*)d_out;                  // [32,512]

    char* ws = (char*)d_ws;
    __bf16* Wt  = (__bf16*)ws;                       // 512 KB
    float*  ait = (float*)(ws + 512 * 1024);         // 256 KB
    float*  a   = (float*)(ws + 768 * 1024);         // 256 KB

    wt_convert_kernel   <<<(D_ * D_) / 256, 256, 0, stream>>>(W, Wt);
    gemm_tanh_dot_kernel<<<M_TOT / MT_,     256, 0, stream>>>(x, Wt, bias, u, ait);
    softmax_t_kernel    <<<B_,              256, 0, stream>>>(ait, a);
    pool_kernel         <<<B_ * 2,          256, 0, stream>>>(x, a, out);
}
